// ARNet_11639361372578
// MI455X (gfx1250) — matrix operand order repaired and checked by reading, not compiled, not run on hardware
//
#include <hip/hip_runtime.h>
#include <hip/hip_bf16.h>

typedef __attribute__((ext_vector_type(16))) _Float16 v16h;
typedef __attribute__((ext_vector_type(8)))  _Float16 v8h;
typedef __attribute__((ext_vector_type(8)))  float    v8f;

#define NA    29
#define DIM   6
#define MDIM  32
#define EIN   13
#define H1    26               // 2*EIN
#define NPAIR (NA * NA)        // 841
#define NTILE ((NPAIR + 15) / 16)  // 53
#define WAVES 4
#define TPB   (WAVES * 32)
#define OUTPA (NA * DIM)       // 174 floats per batch in output

// Fast activations: v_exp_f32 + v_rcp_f32 (no IEEE div expansion).
__device__ __forceinline__ float sigmoidf_(float x) {
  return __builtin_amdgcn_rcpf(1.f + __expf(-x));
}
__device__ __forceinline__ float siluf_(float x) { return x * sigmoidf_(x); }

// k is a compile-time constant after unrolling -> folds to a single value.
__device__ __forceinline__ _Float16 edge_feat_h(int k, _Float16 ci0, _Float16 ci1, _Float16 ci2,
                                                _Float16 cj0, _Float16 cj1, _Float16 cj2,
                                                _Float16 rd) {
  switch (k) {
    case 0: case 3:  return ci0;
    case 1: case 4:  return ci1;
    case 2: case 5:  return ci2;
    case 6: case 9:  return cj0;
    case 7: case 10: return cj1;
    case 8: case 11: return cj2;
    case 12:         return rd;
    default:         return (_Float16)0.f;
  }
}

__global__ __launch_bounds__(TPB) void egnn_wmma_kernel(
    const float* __restrict__ x, const unsigned char* __restrict__ mask,
    const float* __restrict__ W1, const float* __restrict__ b1,
    const float* __restrict__ W2, const float* __restrict__ b2,
    const float* __restrict__ Wg, const float* __restrict__ bg,
    const float* __restrict__ gma, const float* __restrict__ bta,
    const float* __restrict__ Wn1, const float* __restrict__ bn1,
    const float* __restrict__ Wn2, const float* __restrict__ bn2,
    const float* __restrict__ Wm1, const float* __restrict__ bm1,
    const float* __restrict__ Wm2, const float* __restrict__ bm2,
    float* __restrict__ out) {
  __shared__ float cx[NA * 3];
  __shared__ float mi[NA * MDIM];
  __shared__ float foutL[NA * DIM];
  __shared__ float pooledL[DIM];
  __shared__ float hidL[MDIM];
  __shared__ unsigned char mkL[NA + 3];
  __shared__ _Float16 hstage[WAVES][16 * MDIM] __attribute__((aligned(16)));

  const int b    = blockIdx.x;
  const int tid  = threadIdx.x;
  const int lane = tid & 31;
  const int wave = tid >> 5;
  const int nlo  = lane & 15;
  const int hi16 = lane >> 4;    // 0 for lanes 0-15, 1 for lanes 16-31

  // Zero padded output rows 2..28 (elements 12..173); rows 0-1 written later.
  for (int idx = 12 + tid; idx < OUTPA; idx += TPB) out[b * OUTPA + idx] = 0.f;
  for (int idx = tid; idx < NA * MDIM; idx += TPB) mi[idx] = 0.f;
  for (int idx = tid; idx < NA * 3; idx += TPB) cx[idx] = x[b * NA * 3 + idx];
  if (tid < NA) mkL[tid] = mask[b * NA + tid];
  __syncthreads();

  // ---- hoisted per-lane constants: W1/W2 B-tiles (zero-padded), biases, gate ----
  v16h B1t0, B1t1, B2t0, B2t1;
  {
    const int kbase = hi16 ? 8 : 0;  // B layout: lanes 0-15 hold K 0-7,16-23, lanes 16-31 K 8-15,24-31
    #pragma unroll
    for (int e = 0; e < 16; ++e) {
      const int k = kbase + e + ((e < 8) ? 0 : 8);
      const int n0 = nlo, n1 = 16 + nlo;
      B1t0[e] = (_Float16)((k < EIN && n0 < H1) ? W1[k * H1 + n0] : 0.f);
      B1t1[e] = (_Float16)((k < EIN && n1 < H1) ? W1[k * H1 + n1] : 0.f);
      B2t0[e] = (_Float16)((k < H1) ? W2[k * MDIM + n0] : 0.f);
      B2t1[e] = (_Float16)((k < H1) ? W2[k * MDIM + n1] : 0.f);
    }
  }
  v8f c10, c11, c20, c21;
  {
    const float bb10 = b1[nlo];
    const float bb11 = (16 + nlo < H1) ? b1[16 + nlo] : 0.f;
    const float bb20 = b2[nlo];
    const float bb21 = b2[16 + nlo];
    #pragma unroll
    for (int r = 0; r < 8; ++r) { c10[r] = bb10; c11[r] = bb11; c20[r] = bb20; c21[r] = bb21; }
  }
  const float wg0 = Wg[nlo], wg1 = Wg[16 + nlo], bgv = bg[0];
  _Float16* hst = &hstage[wave][0];

  // ---- edge-MLP over 16-pair tiles, 4 WMMAs per tile ----
  for (int t = wave; t < NTILE; t += WAVES) {
    const int p0 = t * 16;
    const int p  = p0 + nlo;               // this lane's A-row pair
    const bool pv = (p < NPAIR);
    const int pi = pv ? (p / NA) : 0;
    const int pj = pv ? (p % NA) : 0;
    const float fi0 = cx[pi * 3 + 0], fi1 = cx[pi * 3 + 1], fi2 = cx[pi * 3 + 2];
    const float fj0 = cx[pj * 3 + 0], fj1 = cx[pj * 3 + 1], fj2 = cx[pj * 3 + 2];
    const float d0 = fi0 - fj0, d1 = fi1 - fj1, d2 = fi2 - fj2;
    const float rdf = d0 * d0 + d1 * d1 + d2 * d2;
    // Convert geometry to f16 once; zero invalid rows here so A assembly is pure b16 selects.
    const _Float16 z  = (_Float16)0.f;
    const _Float16 ci0 = pv ? (_Float16)fi0 : z, ci1 = pv ? (_Float16)fi1 : z,
                   ci2 = pv ? (_Float16)fi2 : z;
    const _Float16 cj0 = pv ? (_Float16)fj0 : z, cj1 = pv ? (_Float16)fj1 : z,
                   cj2 = pv ? (_Float16)fj2 : z;
    const _Float16 rd  = pv ? (_Float16)rdf : z;

    // A1 (16x32 f16): elems 0-7 -> K=kb+e ; elems 8-15 -> K=16+kb+(e-8), kb = hi16*8
    v16h a1;
    #pragma unroll
    for (int e = 0; e < 16; ++e) {
      const int klo = (e < 8) ? e : (e + 8);
      const _Float16 vlo = edge_feat_h(klo,     ci0, ci1, ci2, cj0, cj1, cj2, rd);
      const _Float16 vhi = edge_feat_h(klo + 8, ci0, ci1, ci2, cj0, cj1, cj2, rd);
      a1[e] = hi16 ? vhi : vlo;
    }

    v8f h0 = __builtin_amdgcn_wmma_f32_16x16x32_f16(false, a1, false, B1t0, (short)0, c10, false, false);
    v8f h1 = __builtin_amdgcn_wmma_f32_16x16x32_f16(false, a1, false, B1t1, (short)0, c11, false, false);
    #pragma unroll
    for (int r = 0; r < 8; ++r) { h0[r] = siluf_(h0[r]); h1[r] = siluf_(h1[r]); }

    // Reshape C-layout -> A-layout through LDS (per-wave private slice).
    #pragma unroll
    for (int r = 0; r < 8; ++r) {
      const int m = r + (hi16 << 3);
      hst[m * MDIM + nlo]      = (_Float16)h0[r];
      hst[m * MDIM + 16 + nlo] = (_Float16)h1[r];
    }
    asm volatile("" ::: "memory");  // same-wave DS ops are in-order; block compiler reordering
    v16h a2;
    {
      const v8h* rowp = (const v8h*)(hst + nlo * MDIM);
      const v8h lo  = rowp[hi16];       // halfs [kb, kb+8)
      const v8h hiv = rowp[2 + hi16];   // halfs [16+kb, 16+kb+8)
      #pragma unroll
      for (int e = 0; e < 8; ++e) { a2[e] = lo[e]; a2[8 + e] = hiv[e]; }
    }

    v8f m0 = __builtin_amdgcn_wmma_f32_16x16x32_f16(false, a2, false, B2t0, (short)0, c20, false, false);
    v8f m1 = __builtin_amdgcn_wmma_f32_16x16x32_f16(false, a2, false, B2t1, (short)0, c21, false, false);
    #pragma unroll
    for (int r = 0; r < 8; ++r) { m0[r] = siluf_(m0[r]); m1[r] = siluf_(m1[r]); }

    // Soft edge gate: per-row dot with Wg, reduced within each 16-lane group.
    #pragma unroll
    for (int r = 0; r < 8; ++r) {
      float part = m0[r] * wg0 + m1[r] * wg1;
      part += __shfl_xor(part, 1, 32);
      part += __shfl_xor(part, 2, 32);
      part += __shfl_xor(part, 4, 32);
      part += __shfl_xor(part, 8, 32);
      const float g = sigmoidf_(part + bgv);
      m0[r] *= g; m1[r] *= g;
    }

    // Masked sum over j into m_i (LDS float atomics -> ds_add_f32).
    #pragma unroll
    for (int r = 0; r < 8; ++r) {
      const int m  = r + (hi16 << 3);
      const int pp = p0 + m;
      if (pp < NPAIR) {
        const int ii = pp / NA, jj = pp % NA;
        if (mkL[ii] && mkL[jj]) {
          atomicAdd(&mi[ii * MDIM + nlo],      m0[r]);
          atomicAdd(&mi[ii * MDIM + 16 + nlo], m1[r]);
        }
      }
    }
  }
  __syncthreads();

  // ---- node update + pool + head: tiny, wave 0 only (1 atom per lane) ----
  if (wave == 0) {
    const int a = lane;
    if (a < NA) {
      const float f0 = cx[a * 3 + 0], f1 = cx[a * 3 + 1], f2 = cx[a * 3 + 2];
      const float fv[6] = {f0, f1, f2, f0, f1, f2};
      const float mu = (f0 + f1 + f2) * (1.f / 3.f);
      const float e0 = f0 - mu, e1 = f1 - mu, e2 = f2 - mu;
      const float var = (e0 * e0 + e1 * e1 + e2 * e2) * (1.f / 3.f);
      const float inv = __builtin_amdgcn_rsqf(var + 1e-5f);
      float nrm[6];
      #pragma unroll
      for (int k = 0; k < 6; ++k) nrm[k] = (fv[k] - mu) * inv * gma[k] + bta[k];
      float hdd[12];
      #pragma unroll
      for (int o = 0; o < 12; ++o) {
        float s = bn1[o];
        #pragma unroll
        for (int k = 0; k < 6; ++k) s += nrm[k] * Wn1[k * 12 + o];
        for (int k = 0; k < MDIM; ++k) s += mi[a * MDIM + k] * Wn1[(6 + k) * 12 + o];
        hdd[o] = siluf_(s);
      }
      const float mf = mkL[a] ? 1.f : 0.f;
      #pragma unroll
      for (int o2 = 0; o2 < 6; ++o2) {
        float s = bn2[o2];
        #pragma unroll
        for (int o = 0; o < 12; ++o) s += hdd[o] * Wn2[o * 6 + o2];
        foutL[a * 6 + o2] = (s + fv[o2]) * mf;   // residual, masked for pooling
      }
    }
    asm volatile("" ::: "memory");
    if (lane < 6) {
      float s = 0.f;
      for (int aa = 0; aa < NA; ++aa) s += foutL[aa * 6 + lane];
      pooledL[lane] = s;
    }
    asm volatile("" ::: "memory");
    {
      float s = bm1[lane];
      #pragma unroll
      for (int k = 0; k < 6; ++k) s += pooledL[k] * Wm1[k * MDIM + lane];
      hidL[lane] = fmaxf(s, 0.f);
    }
    asm volatile("" ::: "memory");
    if (lane < 12) {
      float s = bm2[lane];
      for (int h = 0; h < MDIM; ++h) s += hidL[h] * Wm2[h * 12 + lane];
      out[b * OUTPA + lane] = s;   // rows 0..1 of [29,6]
    }
  }
}

extern "C" void kernel_launch(void* const* d_in, const int* in_sizes, int n_in,
                              void* d_out, int out_size, void* d_ws, size_t ws_size,
                              hipStream_t stream) {
  (void)n_in; (void)out_size; (void)d_ws; (void)ws_size;
  const float*         x    = (const float*)d_in[0];
  const unsigned char* mask = (const unsigned char*)d_in[1];
  const float* W1  = (const float*)d_in[2];
  const float* b1  = (const float*)d_in[3];
  const float* W2  = (const float*)d_in[4];
  const float* b2  = (const float*)d_in[5];
  const float* Wg  = (const float*)d_in[6];
  const float* bg  = (const float*)d_in[7];
  const float* gma = (const float*)d_in[8];
  const float* bta = (const float*)d_in[9];
  const float* Wn1 = (const float*)d_in[10];
  const float* bn1 = (const float*)d_in[11];
  const float* Wn2 = (const float*)d_in[12];
  const float* bn2 = (const float*)d_in[13];
  const float* Wm1 = (const float*)d_in[14];
  const float* bm1 = (const float*)d_in[15];
  const float* Wm2 = (const float*)d_in[16];
  const float* bm2 = (const float*)d_in[17];
  float* out = (float*)d_out;

  const int B = in_sizes[0] / (NA * 3);
  egnn_wmma_kernel<<<dim3(B), dim3(TPB), 0, stream>>>(
      x, mask, W1, b1, W2, b2, Wg, bg, gma, bta,
      Wn1, bn1, Wn2, bn2, Wm1, bm1, Wm2, bm2, out);
}
